// TransitionLayerAblation_3332894621737
// MI455X (gfx1250) — matrix operand order repaired and checked by reading, not compiled, not run on hardware
//
#include <hip/hip_runtime.h>
#include <hip/hip_bf16.h>
#include <math.h>

// ---------------------------------------------------------------------------
// GRU cell over N=100000 rows, G=H=256, fused with mask + column-max + time
// branch.  GEMMs run as bf16 WMMA (v_wmma_f32_16x16x32_bf16) with f32
// accumulation; everything else stays fp32.
// ---------------------------------------------------------------------------

typedef __attribute__((ext_vector_type(16))) __bf16 v16bf;
typedef __attribute__((ext_vector_type(8)))  float  v8f;

#define GRAPH   256
#define HIDDEN  256
#define G3      768           // 3*HIDDEN
#define KBLK    8             // 256 / 32
#define BLOCK_ROWS 32         // rows per workgroup (2 M-tiles of 16)
#define WAVES   16            // one wave per 16-wide hidden-column chunk
#define THREADS (WAVES * 32)
// repacked-weight tile: 32x16 bf16 B-fragment = 256 dwords
#define NTILES  (2 * 3 * 16 * KBLK)          // which * gate * colchunk * kblk
#define BW_DWORDS (NTILES * 256)             // 196608 dwords = 768 KB

struct Frag32 { uint4 lo; uint4 hi; };       // 32 bytes -> v16bf
static_assert(sizeof(Frag32) == 32, "frag size");

static __device__ __forceinline__ unsigned pack_bf16x2(float a, float b) {
    unsigned ua = __builtin_bit_cast(unsigned, a);
    unsigned ub = __builtin_bit_cast(unsigned, b);
    // round-to-nearest-even fp32 -> bf16
    unsigned ha = (ua + 0x7FFFu + ((ua >> 16) & 1u)) >> 16;
    unsigned hb = (ub + 0x7FFFu + ((ub >> 16) & 1u)) >> 16;
    return (ha & 0xFFFFu) | (hb << 16);
}

// order-preserving float <-> uint map so we can use integer atomicMax
static __device__ __forceinline__ unsigned fmap(float x) {
    unsigned u = __builtin_bit_cast(unsigned, x);
    return (u & 0x80000000u) ? ~u : (u | 0x80000000u);
}
static __device__ __forceinline__ float funmap(unsigned u) {
    unsigned b = (u & 0x80000000u) ? (u ^ 0x80000000u) : ~u;
    return __builtin_bit_cast(float, b);
}
#define FMAP_NEG_INF 0x007FFFFFu   // fmap(-inf)

static __device__ __forceinline__ float sigmoidf_(float x) {
    return 1.0f / (1.0f + __expf(-x));
}

// ---------------------------------------------------------------------------
// Kernel 1: repack fp32 weights into bf16 B-fragments.
// Tile (which, gate, cc, kb) is a 32(K) x 16(N) slice:
//   N index n -> weight row  r = gate*256 + cc*16 + n
//   K index k -> weight col  k = kb*32 + ...
// Per-lane layout (contiguous K order used here, same as A): lane L: n = L&15, half = L>>4;
// VGPR j holds K = half*16 + 2j, +1.  Stored lane-major: dword off = lane*8+j.
// ---------------------------------------------------------------------------
__global__ void __launch_bounds__(256) repack_weights_kernel(
    const float* __restrict__ wih, const float* __restrict__ whh,
    unsigned* __restrict__ bw)
{
    int o = blockIdx.x * blockDim.x + threadIdx.x;       // [0, BW_DWORDS)
    if (o >= BW_DWORDS) return;
    int j    = o & 7;
    int lane = (o >> 3) & 31;
    int tile = o >> 8;
    int kb   = tile & 7;
    int cc   = (tile >> 3) & 15;
    int d    = tile >> 7;            // 0..5
    int which = d / 3;               // 0 = ih, 1 = hh
    int g     = d % 3;
    int n    = lane & 15;
    int half = lane >> 4;
    int k0   = kb * 32 + half * 16 + j * 2;
    int r    = g * 256 + cc * 16 + n;
    const float* __restrict__ W = which ? whh : wih;
    bw[o] = pack_bf16x2(W[r * GRAPH + k0], W[r * GRAPH + k0 + 1]);
}

// ---------------------------------------------------------------------------
// Kernel 2: init column-max buffer to map(-inf)
// ---------------------------------------------------------------------------
__global__ void init_max_kernel(unsigned* __restrict__ maxbuf) {
    maxbuf[threadIdx.x] = FMAP_NEG_INF;
}

// ---------------------------------------------------------------------------
// A-fragment (16x32 bf16, contiguous K order, same as B): lane L: M = L&15, half = L>>4;
// VGPR j holds K = half*16 + 2j, +1.
// LDS holds packed bf16 pairs, row-major: lds[row*128 + k/2].
// dword index: kb*16 + half*8 + j        (16B aligned blocks)
// ---------------------------------------------------------------------------
static __device__ __forceinline__ v16bf load_a_frag(
    const unsigned* __restrict__ lds, int rowInBlk, int kb, int lane)
{
    int m = lane & 15, half = lane >> 4;
    const unsigned* p = lds + (rowInBlk + m) * 128 + kb * 16 + half * 8;
    Frag32 f;
    f.lo = *(const uint4*)(p);
    f.hi = *(const uint4*)(p + 4);
    return __builtin_bit_cast(v16bf, f);
}

static __device__ __forceinline__ v16bf load_b_frag(
    const unsigned* __restrict__ bw, int which, int g, int cc, int kb, int lane)
{
    int tile = ((which * 3 + g) * 16 + cc) * 8 + kb;
    const unsigned* p = bw + tile * 256 + lane * 8;
    Frag32 f;
    f.lo = *(const uint4*)(p);
    f.hi = *(const uint4*)(p + 4);
    return __builtin_bit_cast(v16bf, f);
}

// ---------------------------------------------------------------------------
// Kernel 3: fused GEMM x2 + GRU + masked scatter + masked column max.
// Workgroup: 16 waves, 32 rows.  Wave w owns hidden columns [16w, 16w+16).
// ---------------------------------------------------------------------------
__global__ void __launch_bounds__(THREADS) gru_main_kernel(
    const float* __restrict__ co,       // [N,256]  x
    const float* __restrict__ hidden,   // [N,256]  h
    const int*   __restrict__ divided,  // [N,3]
    const unsigned* __restrict__ bw,    // repacked bf16 weights
    const float* __restrict__ bias_ih,  // [768]
    const float* __restrict__ bias_hh,  // [768]
    float* __restrict__ hnew,           // [N,256]  (d_out + 256)
    unsigned* __restrict__ maxbuf,      // [256]
    int N)
{
    __shared__ unsigned lds_x[BLOCK_ROWS * 128];  // 16 KB bf16 pairs
    __shared__ unsigned lds_h[BLOCK_ROWS * 128];  // 16 KB

    const int tid  = threadIdx.x;
    const int wave = tid >> 5;
    const int lane = tid & 31;
    const int base = blockIdx.x * BLOCK_ROWS;

    // ---- stage x and h tiles into LDS as packed bf16 --------------------
    const float2* __restrict__ co2 = (const float2*)co;
    const float2* __restrict__ hd2 = (const float2*)hidden;
#pragma unroll
    for (int i = 0; i < (BLOCK_ROWS * 128) / THREADS; ++i) {
        int idx  = tid + i * THREADS;          // pair index
        int row  = idx >> 7;
        int kp   = idx & 127;
        int grow = base + row;
        float2 xv = make_float2(0.f, 0.f), hv = make_float2(0.f, 0.f);
        if (grow < N) {
            xv = co2[grow * 128 + kp];
            hv = hd2[grow * 128 + kp];
        }
        lds_x[idx] = pack_bf16x2(xv.x, xv.y);
        lds_h[idx] = pack_bf16x2(hv.x, hv.y);
    }
    __syncthreads();

    const int cc  = wave;
    const int n   = lane & 15;
    const int col = cc * 16 + n;

    // ---- accumulators, biases folded in ---------------------------------
    v8f accI[3][2], accH[3][2];
#pragma unroll
    for (int g = 0; g < 3; ++g) {
        float bi = bias_ih[g * 256 + col];
        float bh = bias_hh[g * 256 + col];
#pragma unroll
        for (int mt = 0; mt < 2; ++mt)
#pragma unroll
            for (int j = 0; j < 8; ++j) {
                accI[g][mt][j] = bi;
                accH[g][mt][j] = bh;
            }
    }

    // ---- K loop: 8 x (K=32) steps, 12 WMMAs each -------------------------
#pragma unroll
    for (int kb = 0; kb < KBLK; ++kb) {
        v16bf ax0 = load_a_frag(lds_x,  0, kb, lane);
        v16bf ax1 = load_a_frag(lds_x, 16, kb, lane);
        v16bf ah0 = load_a_frag(lds_h,  0, kb, lane);
        v16bf ah1 = load_a_frag(lds_h, 16, kb, lane);
#pragma unroll
        for (int g = 0; g < 3; ++g) {
            v16bf bih = load_b_frag(bw, 0, g, cc, kb, lane);
            v16bf bhh = load_b_frag(bw, 1, g, cc, kb, lane);
            accI[g][0] = __builtin_amdgcn_wmma_f32_16x16x32_bf16(
                false, ax0, false, bih, (short)0, accI[g][0], false, false);
            accI[g][1] = __builtin_amdgcn_wmma_f32_16x16x32_bf16(
                false, ax1, false, bih, (short)0, accI[g][1], false, false);
            accH[g][0] = __builtin_amdgcn_wmma_f32_16x16x32_bf16(
                false, ah0, false, bhh, (short)0, accH[g][0], false, false);
            accH[g][1] = __builtin_amdgcn_wmma_f32_16x16x32_bf16(
                false, ah1, false, bhh, (short)0, accH[g][1], false, false);
        }
    }

    // ---- GRU epilogue.  C layout: lane L -> col = L&15, VGPR j -> row
    //      M = (L>>4)*8 + j ----------------------------------------------
    float colmax = -INFINITY;
#pragma unroll
    for (int mt = 0; mt < 2; ++mt) {
#pragma unroll
        for (int j = 0; j < 8; ++j) {
            int row = base + mt * 16 + (lane >> 4) * 8 + j;
            if (row < N) {
                float ir = accI[0][mt][j], iz = accI[1][mt][j], in_ = accI[2][mt][j];
                float hr = accH[0][mt][j], hz = accH[1][mt][j], hn  = accH[2][mt][j];
                float r  = sigmoidf_(ir + hr);
                float z  = sigmoidf_(iz + hz);
                float nn = tanhf(in_ + r * hn);
                float hp = hidden[row * HIDDEN + col];
                float o  = (1.0f - z) * nn + z * hp;
                bool  m  = (divided[row * 3 + 0] > 0) |
                           (divided[row * 3 + 1] > 0) |
                           (divided[row * 3 + 2] > 0);
                hnew[row * HIDDEN + col] = m ? o : 0.0f;
                colmax = fmaxf(colmax, m ? o : -INFINITY);
            }
        }
    }
    // lanes L and L+16 hold the same column -> combine halves
    float other = __shfl_xor(colmax, 16);
    colmax = fmaxf(colmax, other);
    if (lane < 16) atomicMax(&maxbuf[col], fmap(colmax));
}

// ---------------------------------------------------------------------------
// Kernel 4: finalize — output[c] = max[c] + tanh(time_w[c]/log(interval+e)+b)
// ---------------------------------------------------------------------------
__global__ void finalize_kernel(
    const float* __restrict__ interval,
    const float* __restrict__ time_w, const float* __restrict__ time_b,
    const unsigned* __restrict__ maxbuf, float* __restrict__ out)
{
    int c = threadIdx.x;                         // 256 threads
    float inv = 1.0f / logf(interval[0] + 2.71828182845904523536f);
    float tf  = tanhf(inv * time_w[c] + time_b[c]);
    out[c] = funmap(maxbuf[c]) + tf;
}

// ---------------------------------------------------------------------------
extern "C" void kernel_launch(void* const* d_in, const int* in_sizes, int n_in,
                              void* d_out, int out_size, void* d_ws, size_t ws_size,
                              hipStream_t stream)
{
    const float* interval = (const float*)d_in[0];
    const float* co       = (const float*)d_in[2];
    const int*   divided  = (const int*)  d_in[3];
    const float* hidden   = (const float*)d_in[7];
    const float* wih      = (const float*)d_in[8];
    const float* whh      = (const float*)d_in[9];
    const float* bih      = (const float*)d_in[10];
    const float* bhh      = (const float*)d_in[11];
    const float* tw       = (const float*)d_in[12];
    const float* tb       = (const float*)d_in[13];

    const int N = in_sizes[2] / GRAPH;           // 100000

    unsigned* bw     = (unsigned*)d_ws;          // BW_DWORDS dwords
    unsigned* maxbuf = bw + BW_DWORDS;           // 256 dwords

    float* out  = (float*)d_out;                 // [0:256) = output+time
    float* hnew = out + HIDDEN;                  // [256: 256 + N*256)

    repack_weights_kernel<<<BW_DWORDS / 256, 256, 0, stream>>>(wih, whh, bw);
    init_max_kernel<<<1, 256, 0, stream>>>(maxbuf);

    int nblocks = (N + BLOCK_ROWS - 1) / BLOCK_ROWS;
    gru_main_kernel<<<nblocks, THREADS, 0, stream>>>(
        co, hidden, divided, bw, bih, bhh, hnew, maxbuf, N);

    finalize_kernel<<<1, 256, 0, stream>>>(interval, tw, tb, maxbuf, out);
}
